// BDHModel_858993459838
// MI455X (gfx1250) — 16-bit and 8-bit matrix operand order repaired and checked by reading, compiled, not run on hardware
//
#include <hip/hip_runtime.h>
#include <hip/hip_bf16.h>
#include <math.h>


typedef __attribute__((ext_vector_type(8)))  int      v8i;
typedef __attribute__((ext_vector_type(8)))  float    v8f;
typedef __attribute__((ext_vector_type(8)))  _Float16 v8h;
typedef __attribute__((ext_vector_type(16))) _Float16 v16h;

#define NFEAT 2048
#define NTOK  2048
#define TM    2047
#define KTOP  40
#define LRATE 0.005f
#define EPSV  1e-8f

#define OFF_ACTS 0ull
#define OFF_CNT  4194304ull
#define OFF_G    4202496ull
#define OFF_QH   20979712ull
#define OFF_MMT  29368320ull
#define OFF_RT   37756928ull

__global__ __launch_bounds__(256) void act_kernel(const int* __restrict__ tokens,
                                                  const float* __restrict__ proj,
                                                  unsigned char* __restrict__ acts,
                                                  int* __restrict__ cnt)
{
    const int t   = blockIdx.x;
    const int tid = threadIdx.x;
    const float* row = proj + (size_t)tokens[t] * NFEAT;

    unsigned int u[8];
#pragma unroll
    for (int i = 0; i < 8; ++i) {
        unsigned int b = __float_as_uint(row[tid + 256 * i]);
        u[i] = (b & 0x80000000u) ? ~b : (b | 0x80000000u);
    }

    __shared__ int hist[256];
    __shared__ unsigned int s_prefix;
    __shared__ int s_k;
    __shared__ int s_cnt;
    if (tid == 0) { s_prefix = 0u; s_k = KTOP; s_cnt = 0; }
    __syncthreads();

    for (int pass = 3; pass >= 0; --pass) {
        hist[tid] = 0;
        __syncthreads();
        const unsigned int prefix = s_prefix;
        const int shift = pass * 8;
        const unsigned int himask = (pass == 3) ? 0u : (0xFFFFFFFFu << (shift + 8));
#pragma unroll
        for (int i = 0; i < 8; ++i) {
            if ((u[i] & himask) == (prefix & himask))
                atomicAdd(&hist[(u[i] >> shift) & 0xFF], 1);
        }
        __syncthreads();
        if (tid == 0) {
            int k = s_k, cum = 0, b = 255;
            for (; b > 0; --b) { cum += hist[b]; if (cum >= k) break; }
            if (cum < k) cum += hist[0];
            s_k = k - (cum - hist[b]);
            s_prefix = prefix | ((unsigned int)b << shift);
        }
        __syncthreads();
    }

    const unsigned int thr = s_prefix;
    int local = 0;
#pragma unroll
    for (int i = 0; i < 8; ++i) {
        unsigned char a = (u[i] >= thr) ? 1 : 0;
        acts[(size_t)t * NFEAT + tid + 256 * i] = a;
        local += a;
    }
    atomicAdd(&s_cnt, local);
    __syncthreads();
    if (tid == 0) cnt[t] = s_cnt;
}

struct FragI8 { unsigned long long q[4]; };
__device__ __forceinline__ v8i frag_as_v8i(const FragI8& f)
{
    union { FragI8 f; v8i v; } u; u.f = f; return u.v;
}
__device__ __forceinline__ void load_a_i8(FragI8& f, const unsigned char* p)
{
    const unsigned long long* q = (const unsigned long long*)p;
#pragma unroll
    for (int v = 0; v < 4; ++v) f.q[v] = q[v];
}
__device__ __forceinline__ void load_b_i8(FragI8& f, const unsigned char* p)
{
    const unsigned long long* q = (const unsigned long long*)p;
#pragma unroll
    for (int v = 0; v < 4; ++v) f.q[v] = q[v];
}

__global__ __launch_bounds__(256) void gram_iu8_kernel(const unsigned char* __restrict__ acts,
                                                       int* __restrict__ G)
{
    const int lane  = threadIdx.x & 31;
    const int wave  = threadIdx.x >> 5;
    const int strip = blockIdx.x * 8 + wave;
    const int mt = strip >> 5;
    const int ns = strip & 31;
    const int hi = lane >> 4;
    const int lo = lane & 15;

    v8i acc0 = {}, acc1 = {}, acc2 = {}, acc3 = {};
    const unsigned char* ap  = acts + (size_t)(mt * 16 + lo) * NFEAT + 32 * hi;
    const unsigned char* b0p = acts + (size_t)(ns * 64 +  0 + lo) * NFEAT + 32 * hi;
    const unsigned char* b1p = acts + (size_t)(ns * 64 + 16 + lo) * NFEAT + 32 * hi;
    const unsigned char* b2p = acts + (size_t)(ns * 64 + 32 + lo) * NFEAT + 32 * hi;
    const unsigned char* b3p = acts + (size_t)(ns * 64 + 48 + lo) * NFEAT + 32 * hi;

    for (int k0 = 0; k0 < NFEAT; k0 += 64) {
        __builtin_prefetch(ap + k0 + 128, 0, 3);
        FragI8 a, b0, b1, b2, b3;
        load_a_i8(a,  ap  + k0);
        load_b_i8(b0, b0p + k0);
        load_b_i8(b1, b1p + k0);
        load_b_i8(b2, b2p + k0);
        load_b_i8(b3, b3p + k0);
        acc0 = __builtin_amdgcn_wmma_i32_16x16x64_iu8(false, frag_as_v8i(a), false, frag_as_v8i(b0), acc0, false, false);
        acc1 = __builtin_amdgcn_wmma_i32_16x16x64_iu8(false, frag_as_v8i(a), false, frag_as_v8i(b1), acc1, false, false);
        acc2 = __builtin_amdgcn_wmma_i32_16x16x64_iu8(false, frag_as_v8i(a), false, frag_as_v8i(b2), acc2, false, false);
        acc3 = __builtin_amdgcn_wmma_i32_16x16x64_iu8(false, frag_as_v8i(a), false, frag_as_v8i(b3), acc3, false, false);
    }

    int* grow = G + (size_t)(mt * 16 + 8 * hi) * NFEAT + ns * 64 + lo;
#pragma unroll
    for (int v = 0; v < 8; ++v) {
        grow[(size_t)v * NFEAT +  0] = acc0[v];
        grow[(size_t)v * NFEAT + 16] = acc1[v];
        grow[(size_t)v * NFEAT + 32] = acc2[v];
        grow[(size_t)v * NFEAT + 48] = acc3[v];
    }
}

__global__ __launch_bounds__(256) void build_f16_kernel(const int* __restrict__ G,
                                                        _Float16* __restrict__ Qh,
                                                        _Float16* __restrict__ MmT)
{
    const size_t idx = (size_t)blockIdx.x * 256 + threadIdx.x;
    const int i = (int)(idx >> 11);
    const int j = (int)(idx & 2047);
    Qh[idx] = (i < TM && j < TM) ? (_Float16)(float)G[idx] : (_Float16)0.0f;
    MmT[idx] = (j < i && i < TM) ? (_Float16)(float)G[(size_t)(j + 1) * NFEAT + i]
                                 : (_Float16)0.0f;
}

__global__ __launch_bounds__(256) void qform_f16_kernel(const _Float16* __restrict__ Qh,
                                                        const _Float16* __restrict__ MmT,
                                                        float* __restrict__ RT)
{
    const int lane  = threadIdx.x & 31;
    const int wave  = threadIdx.x >> 5;
    const int strip = blockIdx.x * 8 + wave;
    const int st = strip >> 5;
    const int ts = strip & 31;
    const int hi = lane >> 4;
    const int lo = lane & 15;

    v8f acc0 = {}, acc1 = {}, acc2 = {}, acc3 = {};
    const _Float16* arow = Qh  + (size_t)(st * 16 + lo) * NFEAT + 16 * hi;
    const _Float16* b0p  = MmT + (size_t)(ts * 64 +  0 + lo) * NFEAT + 16 * hi;
    const _Float16* b1p  = MmT + (size_t)(ts * 64 + 16 + lo) * NFEAT + 16 * hi;
    const _Float16* b2p  = MmT + (size_t)(ts * 64 + 32 + lo) * NFEAT + 16 * hi;
    const _Float16* b3p  = MmT + (size_t)(ts * 64 + 48 + lo) * NFEAT + 16 * hi;

    for (int k0 = 0; k0 < NFEAT; k0 += 32) {
        __builtin_prefetch(arow + k0 + 64, 0, 3);
        v8h aL = *(const v8h*)(arow + k0);
        v8h aH = *(const v8h*)(arow + k0 + 8);
        v16h aa = __builtin_shufflevector(aL, aH,
                      0,1,2,3,4,5,6,7,8,9,10,11,12,13,14,15);
        v16h bb0 = *(const v16h*)(b0p + k0);
        v16h bb1 = *(const v16h*)(b1p + k0);
        v16h bb2 = *(const v16h*)(b2p + k0);
        v16h bb3 = *(const v16h*)(b3p + k0);
        acc0 = __builtin_amdgcn_wmma_f32_16x16x32_f16(false, aa, false, bb0, (short)0, acc0, false, false);
        acc1 = __builtin_amdgcn_wmma_f32_16x16x32_f16(false, aa, false, bb1, (short)0, acc1, false, false);
        acc2 = __builtin_amdgcn_wmma_f32_16x16x32_f16(false, aa, false, bb2, (short)0, acc2, false, false);
        acc3 = __builtin_amdgcn_wmma_f32_16x16x32_f16(false, aa, false, bb3, (short)0, acc3, false, false);
    }

    const int srow0 = st * 16 + 8 * hi;
#pragma unroll
    for (int v = 0; v < 8; ++v) {
        RT[(size_t)(ts * 64 +  0 + lo) * NFEAT + srow0 + v] = acc0[v];
        RT[(size_t)(ts * 64 + 16 + lo) * NFEAT + srow0 + v] = acc1[v];
        RT[(size_t)(ts * 64 + 32 + lo) * NFEAT + srow0 + v] = acc2[v];
        RT[(size_t)(ts * 64 + 48 + lo) * NFEAT + srow0 + v] = acc3[v];
    }
}

__global__ __launch_bounds__(256) void tension_kernel(const int* __restrict__ G,
                                                      const _Float16* __restrict__ MmT,
                                                      const float* __restrict__ RT,
                                                      const int* __restrict__ cnt,
                                                      float* __restrict__ out)
{
    const int t   = blockIdx.x;
    const int tid = threadIdx.x;
    float qs = 0.f, nm = 0.f;
    for (int s = tid; s < NFEAT; s += 256) {
        const float m = (float)MmT[(size_t)t * NFEAT + s];
        qs += m * RT[(size_t)t * NFEAT + s];
        if (s < t)
            nm += (float)G[(size_t)(s + 1) * NFEAT + t] *
                  (float)G[(size_t)(t + 1) * NFEAT + s];
    }
    __shared__ float rq[256], rn[256];
    rq[tid] = qs; rn[tid] = nm;
    __syncthreads();
    for (int w = 128; w > 0; w >>= 1) {
        if (tid < w) { rq[tid] += rq[tid + w]; rn[tid] += rn[tid + w]; }
        __syncthreads();
    }
    if (tid == 0) {
        const float q = rq[0], n = rn[0];
        float overlap = 0.f;
        if (q > 0.f) {
            const float pnorm = LRATE * sqrtf(q);
            const float xnn   = sqrtf((float)cnt[t + 1]);
            overlap = (LRATE * n) / (pnorm * xnn + EPSV);
        }
        out[t] = 1.f - overlap;
    }
}

extern "C" void kernel_launch(void* const* d_in, const int* in_sizes, int n_in,
                              void* d_out, int out_size, void* d_ws, size_t ws_size,
                              hipStream_t stream)
{
    const int*   tokens = (const int*)d_in[0];
    const float* proj   = (const float*)d_in[1];

    char* ws = (char*)d_ws;
    unsigned char* acts = (unsigned char*)(ws + OFF_ACTS);
    int*           cnt  = (int*)(ws + OFF_CNT);
    int*           G    = (int*)(ws + OFF_G);
    _Float16*      Qh   = (_Float16*)(ws + OFF_QH);
    _Float16*      MmT  = (_Float16*)(ws + OFF_MMT);
    float*         RT   = (float*)(ws + OFF_RT);
    float*         out  = (float*)d_out;

    act_kernel      <<<dim3(NTOK),            dim3(256), 0, stream>>>(tokens, proj, acts, cnt);
    gram_iu8_kernel <<<dim3(512),             dim3(256), 0, stream>>>(acts, G);
    build_f16_kernel<<<dim3(NFEAT*NFEAT/256), dim3(256), 0, stream>>>(G, Qh, MmT);
    qform_f16_kernel<<<dim3(512),             dim3(256), 0, stream>>>(Qh, MmT, RT);
    tension_kernel  <<<dim3(TM),              dim3(256), 0, stream>>>(G, MmT, RT, cnt, out);
}
